// GCN_ConvE_56779467653350
// MI455X (gfx1250) — matrix operand order repaired and checked by reading, not compiled, not run on hardware
//
#include <hip/hip_runtime.h>
#include <hip/hip_bf16.h>
#include <math.h>

// ---------------- constants from the reference ----------------
#define N_ENT   100000
#define D       200
#define KPAD    224        // 7 * 32
#define KT_D    7
#define NT_D    13         // 13 * 16 = 208 >= 200
#define NREL2   500        // 2 * NUM_REL
#define E_TOT   400000
#define E_HALF  200000
#define BATCH   1024
#define NFILT   200
#define FLATK   39200      // 200*14*14, == 1225 * 32 exactly
#define KT_FC   1225
#define NT_ENT  6250       // 100000 / 16 exactly
#define EPSBN   1e-5f
#define MBLK    64         // rows of A per block (4 m-tiles of 16)

typedef _Float16 v16h __attribute__((ext_vector_type(16)));
typedef _Float16 h8v  __attribute__((ext_vector_type(8)));
typedef float    v8f  __attribute__((ext_vector_type(8)));

union AFrag { v16h v; h8v h[2]; };

static __device__ inline v8f wmma16(v16h a, v16h b, v8f c) {
  // D(f32 16x16) = A(f16 16x32) * B(f16 32x16) + C
  return __builtin_amdgcn_wmma_f32_16x16x32_f16(false, a, false, b, (short)0, c, false, false);
}

// ---------------- weight packing: W[K x N] f32 -> per-fragment f16 ----------------
// Fragment (kt,nt): lane l holds column n = nt*16 + (l&15), K = kt*32 + (e<8 ? (l>>4)*8 + e : 16 + (l>>4)*8 + e-8).
__global__ void pack_w_kernel(const float* __restrict__ W, int K, int N, int KT, int NT,
                              _Float16* __restrict__ out) {
  int wave = threadIdx.x >> 5, lane = threadIdx.x & 31;
  int frag = blockIdx.x * (blockDim.x >> 5) + wave;
  if (frag >= KT * NT) return;
  int kt = frag / NT, nt = frag - kt * NT;
  int n = nt * 16 + (lane & 15), hh = lane >> 4;
  _Float16* dst = out + (size_t)frag * 512 + lane * 16;
#pragma unroll
  for (int e = 0; e < 16; ++e) {
    int k = kt * 32 + (e < 8 ? hh * 8 + e : 16 + hh * 8 + (e - 8));
    float v = (k < K && n < N) ? W[(size_t)k * N + n] : 0.f;
    dst[e] = (_Float16)v;
  }
}

// Pack B = X^T where X is [NE x Dd] row-major (logits GEMM: columns are entities).
__global__ void pack_xT_kernel(const float* __restrict__ X, int NE, int Dd, int KT, int NT,
                               _Float16* __restrict__ out) {
  int wave = threadIdx.x >> 5, lane = threadIdx.x & 31;
  int frag = blockIdx.x * (blockDim.x >> 5) + wave;
  if (frag >= KT * NT) return;
  int kt = frag / NT, nt = frag - kt * NT;
  int ent = nt * 16 + (lane & 15), hh = lane >> 4;
  const float* src = X + (size_t)ent * Dd;
  _Float16* dst = out + (size_t)frag * 512 + lane * 16;
#pragma unroll
  for (int e = 0; e < 16; ++e) {
    int k = kt * 32 + (e < 8 ? hh * 8 + e : 16 + hh * 8 + (e - 8));
    float v = (ent < NE && k < Dd) ? src[k] : 0.f;
    dst[e] = (_Float16)v;
  }
}

// ------- edge message GEMM + scatter: agg[ed] += norm * ((x[es]*r[et]) @ W) -------
// 64 edges per block, 8 waves; each B fragment feeds 4 WMMAs (4 m-tiles).
__global__ __launch_bounds__(256) void edge_msg_kernel(
    const float* __restrict__ x, const float* __restrict__ r,
    const int* __restrict__ es, const int* __restrict__ ed,
    const int* __restrict__ et, const float* __restrict__ norm,
    const _Float16* __restrict__ wIn, const _Float16* __restrict__ wOut,
    float* __restrict__ agg) {
  __shared__ int sSrc[MBLK], sDst[MBLK], sTyp[MBLK];
  __shared__ float sNrm[MBLK];
  __shared__ _Float16 sA[MBLK * KPAD] __attribute__((aligned(16)));
  int tid = threadIdx.x;
  int ebase = blockIdx.x * MBLK;
  if (tid < MBLK) {
    sSrc[tid] = es[ebase + tid];
    sDst[tid] = ed[ebase + tid];
    sTyp[tid] = et[ebase + tid];
    sNrm[tid] = norm[ebase + tid];
  }
  __syncthreads();
  for (int idx = tid; idx < MBLK * KPAD; idx += 256) {
    int i = idx / KPAD, k = idx - i * KPAD;
    float v = 0.f;
    if (k < D) v = x[(size_t)sSrc[i] * D + k] * r[(size_t)sTyp[i] * D + k];
    sA[idx] = (_Float16)v;
  }
  __syncthreads();
  const _Float16* Bp = (ebase >= E_HALF) ? wOut : wIn;  // E_HALF % MBLK == 0, no straddle
  int wave = tid >> 5, lane = tid & 31;
  const _Float16* ap0 = &sA[(lane & 15) * KPAD + (lane >> 4) * 8];
  for (int nt = wave; nt < NT_D; nt += 8) {
    v8f acc[4] = {};
#pragma unroll
    for (int kt = 0; kt < KT_D; ++kt) {
      v16h b = *(const v16h*)(Bp + ((size_t)kt * NT_D + nt) * 512 + lane * 16);
#pragma unroll
      for (int mt = 0; mt < 4; ++mt) {
        AFrag a;
        const _Float16* ap = ap0 + mt * 16 * KPAD + kt * 32;
        a.h[0] = *(const h8v*)ap;
        a.h[1] = *(const h8v*)(ap + 16);
        acc[mt] = wmma16(a.v, b, acc[mt]);
      }
    }
    int n = nt * 16 + (lane & 15);
    if (n < D) {
#pragma unroll
      for (int mt = 0; mt < 4; ++mt) {
#pragma unroll
        for (int rr = 0; rr < 8; ++rr) {
          int m = mt * 16 + (lane >> 4) * 8 + rr;
          atomicAdd(&agg[(size_t)sDst[m] * D + n], acc[mt][rr] * sNrm[m]);
        }
      }
    }
  }
}

// ---------------- staged-A GEMM (K=200 fixed): out = epilogue(A' @ Bpack) ----------------
// A'[m][k] = A[m][k] * (kscale ? kscale[k] : 1). If addsrc: out = (addsrc + gemm)*scale + bias.
// 64 rows per block, 8 waves, 4 m-tiles per B fragment.
__global__ __launch_bounds__(256) void gemm_stageA_kernel(
    const float* __restrict__ A, int M, int lda,
    const float* __restrict__ kscale,
    const _Float16* __restrict__ Bpack, int NT,
    const float* __restrict__ addsrc,
    const float* __restrict__ bias, float scale,
    float* __restrict__ out, int ldo, int ncols) {
  __shared__ _Float16 sA[MBLK * KPAD] __attribute__((aligned(16)));
  int tid = threadIdx.x;
  int mbase = blockIdx.x * MBLK;
  for (int idx = tid; idx < MBLK * KPAD; idx += 256) {
    int i = idx / KPAD, k = idx - i * KPAD;
    int m = mbase + i;
    float v = 0.f;
    if (m < M && k < D) {
      v = A[(size_t)m * lda + k];
      if (kscale) v *= kscale[k];
    }
    sA[idx] = (_Float16)v;
  }
  __syncthreads();
  int wave = tid >> 5, lane = tid & 31;
  const _Float16* ap0 = &sA[(lane & 15) * KPAD + (lane >> 4) * 8];
  for (int nt = wave; nt < NT; nt += 8) {
    v8f acc[4] = {};
#pragma unroll
    for (int kt = 0; kt < KT_D; ++kt) {
      v16h b = *(const v16h*)(Bpack + ((size_t)kt * NT + nt) * 512 + lane * 16);
#pragma unroll
      for (int mt = 0; mt < 4; ++mt) {
        AFrag a;
        const _Float16* ap = ap0 + mt * 16 * KPAD + kt * 32;
        a.h[0] = *(const h8v*)ap;
        a.h[1] = *(const h8v*)(ap + 16);
        acc[mt] = wmma16(a.v, b, acc[mt]);
      }
    }
    int n = nt * 16 + (lane & 15);
    if (n < ncols) {
#pragma unroll
      for (int mt = 0; mt < 4; ++mt) {
#pragma unroll
        for (int rr = 0; rr < 8; ++rr) {
          int m = mbase + mt * 16 + (lane >> 4) * 8 + rr;
          if (m < M) {
            float v = acc[mt][rr];
            if (addsrc) v = (addsrc[(size_t)m * ldo + n] + v) * scale + bias[n];
            out[(size_t)m * ldo + n] = v;
          }
        }
      }
    }
  }
}

// ---------------- direct f16-A GEMM (long K): fc and logits ----------------
// 64 rows per block (M must be a multiple of 64), 8 waves over nt, 4 m-tiles.
__global__ __launch_bounds__(256) void gemm_f16A_kernel(
    const _Float16* __restrict__ A, int lda, int KT,
    const _Float16* __restrict__ Bpack, int NT,
    const float* __restrict__ bias,
    float* __restrict__ out, int ldo, int ncols, int do_sigmoid) {
  int wave = threadIdx.x >> 5, lane = threadIdx.x & 31;
  int nt = blockIdx.y * 8 + wave;
  if (nt >= NT) return;
  int mbase = blockIdx.x * MBLK;
  const _Float16* arow = A + (size_t)(mbase + (lane & 15)) * lda + (lane >> 4) * 8;
  const _Float16* bcol = Bpack + (size_t)nt * 512 + lane * 16;
  v8f acc[4] = {};
  for (int kt = 0; kt < KT; ++kt) {
    v16h b = *(const v16h*)(bcol + (size_t)kt * NT * 512);
    if (kt + 1 < KT) __builtin_prefetch(bcol + (size_t)(kt + 1) * NT * 512, 0, 3);
#pragma unroll
    for (int mt = 0; mt < 4; ++mt) {
      AFrag a;
      const _Float16* ap = arow + (size_t)(mt * 16) * lda + kt * 32;
      a.h[0] = *(const h8v*)ap;
      a.h[1] = *(const h8v*)(ap + 16);
      acc[mt] = wmma16(a.v, b, acc[mt]);
    }
  }
  int n = nt * 16 + (lane & 15);
  if (n >= ncols) return;
  float bval = bias ? bias[n] : 0.f;
#pragma unroll
  for (int mt = 0; mt < 4; ++mt) {
#pragma unroll
    for (int rr = 0; rr < 8; ++rr) {
      int m = mbase + mt * 16 + (lane >> 4) * 8 + rr;
      float v = acc[mt][rr] + bval;
      if (do_sigmoid) v = 1.f / (1.f + __expf(-v));
      out[(size_t)m * ldo + n] = v;
    }
  }
}

// ---------------- batchnorm helpers ----------------
__global__ void colstats_kernel(const float* __restrict__ H, int M, int N,
                                float* __restrict__ mean, float* __restrict__ rstd) {
  __shared__ float s1[256], s2[256];
  int n = blockIdx.x, tid = threadIdx.x;
  float a = 0.f, b = 0.f;
  for (int m = tid; m < M; m += 256) {
    float v = H[(size_t)m * N + n];
    a += v; b += v * v;
  }
  s1[tid] = a; s2[tid] = b;
  __syncthreads();
  for (int s = 128; s > 0; s >>= 1) {
    if (tid < s) { s1[tid] += s1[tid + s]; s2[tid] += s2[tid + s]; }
    __syncthreads();
  }
  if (tid == 0) {
    float mu = s1[0] / (float)M;
    float var = s2[0] / (float)M - mu * mu;
    mean[n] = mu;
    rstd[n] = rsqrtf(var + EPSBN);
  }
}

__global__ void bn_tanh_kernel(const float* __restrict__ H, const float* __restrict__ mean,
                               const float* __restrict__ rstd, float* __restrict__ out,
                               size_t total, int N) {
  for (size_t idx = blockIdx.x * 256ULL + threadIdx.x; idx < total;
       idx += (size_t)gridDim.x * 256) {
    int n = (int)(idx % (size_t)N);
    out[idx] = tanhf((H[idx] - mean[n]) * rstd[n]);
  }
}

__global__ void gather_stack_kernel(const float* __restrict__ x2, const float* __restrict__ r2,
                                    const int* __restrict__ subj, const int* __restrict__ rel,
                                    float* __restrict__ stack) {
  int idx = blockIdx.x * 256 + threadIdx.x;
  if (idx >= BATCH * 2 * D) return;
  int b = idx / (2 * D), j = idx - b * 2 * D;
  float v = (j < D) ? x2[(size_t)subj[b] * D + j] : r2[(size_t)rel[b] * D + (j - D)];
  stack[idx] = v;
}

__global__ void bn0_stats_kernel(const float* __restrict__ stack, float* __restrict__ st) {
  __shared__ float s1[1024], s2[1024];
  int tid = threadIdx.x;
  float a = 0.f, b = 0.f;
  const int total = BATCH * 2 * D;
  for (int i = tid; i < total; i += 1024) { float v = stack[i]; a += v; b += v * v; }
  s1[tid] = a; s2[tid] = b;
  __syncthreads();
  for (int s = 512; s > 0; s >>= 1) {
    if (tid < s) { s1[tid] += s1[tid + s]; s2[tid] += s2[tid + s]; }
    __syncthreads();
  }
  if (tid == 0) {
    float mu = s1[0] / (float)total;
    float var = s2[0] / (float)total - mu * mu;
    st[0] = mu; st[1] = rsqrtf(var + EPSBN);
  }
}

// ---------------- ConvE 7x7 valid conv over 20x20 image, 200 filters ----------------
__global__ void conv_kernel(const float* __restrict__ stack, const float* __restrict__ st,
                            const float* __restrict__ ck, const float* __restrict__ cb,
                            _Float16* __restrict__ act) {
  __shared__ float img[400];
  __shared__ float kern[NFILT * 49];
  int tid = threadIdx.x, b = blockIdx.x;
  float mu = st[0], rs = st[1];
  for (int i = tid; i < 400; i += 256) img[i] = (stack[b * 400 + i] - mu) * rs;
  for (int i = tid; i < NFILT * 49; i += 256) kern[i] = ck[i];
  __syncthreads();
  for (int o = tid; o < FLATK; o += 256) {
    int f = o / 196, yx = o - f * 196;
    int y = yx / 14, xx = yx - y * 14;
    float acc = cb[f];
    const float* kf = &kern[f * 49];
#pragma unroll
    for (int ky = 0; ky < 7; ++ky)
#pragma unroll
      for (int kx = 0; kx < 7; ++kx)
        acc += img[(y + ky) * 20 + (xx + kx)] * kf[ky * 7 + kx];
    act[(size_t)b * FLATK + o] = (_Float16)acc;
  }
}

__global__ void chanstats_kernel(const _Float16* __restrict__ act,
                                 float* __restrict__ mean, float* __restrict__ rstd) {
  __shared__ float s1[256], s2[256];
  int f = blockIdx.x, tid = threadIdx.x;
  float a = 0.f, bb = 0.f;
  for (int idx = tid; idx < BATCH * 196; idx += 256) {
    int b = idx / 196, i = idx - b * 196;
    float v = (float)act[(size_t)b * FLATK + f * 196 + i];
    a += v; bb += v * v;
  }
  s1[tid] = a; s2[tid] = bb;
  __syncthreads();
  for (int s = 128; s > 0; s >>= 1) {
    if (tid < s) { s1[tid] += s1[tid + s]; s2[tid] += s2[tid + s]; }
    __syncthreads();
  }
  if (tid == 0) {
    float cnt = (float)(BATCH * 196);
    float mu = s1[0] / cnt;
    float var = s2[0] / cnt - mu * mu;
    mean[f] = mu; rstd[f] = rsqrtf(var + EPSBN);
  }
}

__global__ void bn1_relu_kernel(_Float16* __restrict__ act, const float* __restrict__ mean,
                                const float* __restrict__ rstd) {
  const size_t total = (size_t)BATCH * FLATK;
  for (size_t idx = blockIdx.x * 256ULL + threadIdx.x; idx < total;
       idx += (size_t)gridDim.x * 256) {
    int o = (int)(idx % (size_t)FLATK);
    int f = o / 196;
    float v = ((float)act[idx] - mean[f]) * rstd[f];
    act[idx] = (_Float16)(v > 0.f ? v : 0.f);
  }
}

__global__ void bn2_relu_pack_kernel(const float* __restrict__ h, const float* __restrict__ mean,
                                     const float* __restrict__ rstd, _Float16* __restrict__ out) {
  int idx = blockIdx.x * 256 + threadIdx.x;
  if (idx >= BATCH * KPAD) return;
  int m = idx / KPAD, k = idx - m * KPAD;
  float v = 0.f;
  if (k < D) {
    v = (h[m * D + k] - mean[k]) * rstd[k];
    v = v > 0.f ? v : 0.f;
  }
  out[idx] = (_Float16)v;
}

// ---------------- launcher ----------------
extern "C" void kernel_launch(void* const* d_in, const int* in_sizes, int n_in,
                              void* d_out, int out_size, void* d_ws, size_t ws_size,
                              hipStream_t stream) {
  (void)in_sizes; (void)n_in; (void)out_size; (void)ws_size;
  const float* init_embed = (const float*)d_in[0];
  const float* init_rel   = (const float*)d_in[1];
  const float* w1_in      = (const float*)d_in[2];
  const float* w1_out     = (const float*)d_in[3];
  const float* w1_loop    = (const float*)d_in[4];
  const float* w1_rel     = (const float*)d_in[5];
  const float* w1_lrel    = (const float*)d_in[6];
  const float* b1         = (const float*)d_in[7];
  const float* w2_in      = (const float*)d_in[8];
  const float* w2_out     = (const float*)d_in[9];
  const float* w2_loop    = (const float*)d_in[10];
  const float* w2_rel     = (const float*)d_in[11];
  const float* w2_lrel    = (const float*)d_in[12];
  const float* b2         = (const float*)d_in[13];
  const float* conv_k     = (const float*)d_in[14];
  const float* conv_b     = (const float*)d_in[15];
  const float* fc_w       = (const float*)d_in[16];
  const float* fc_b       = (const float*)d_in[17];
  const float* ent_bias   = (const float*)d_in[18];
  const float* edge_norm  = (const float*)d_in[19];
  const int*   edge_src   = (const int*)d_in[20];
  const int*   edge_dst   = (const int*)d_in[21];
  const int*   edge_type  = (const int*)d_in[22];
  const int*   subj       = (const int*)d_in[23];
  const int*   rel        = (const int*)d_in[24];
  float* out = (float*)d_out;

  // ---- workspace carve-out (256B aligned) ----
  char* base = (char*)d_ws;
  size_t off = 0;
  auto take = [&](size_t bytes) -> void* {
    size_t cur = (off + 255) & ~(size_t)255;
    off = cur + bytes;
    return (void*)(base + cur);
  };
  float* agg = (float*)take((size_t)N_ENT * D * 4);   // aggregation / pre-BN (in-place)
  float* x1  = (float*)take((size_t)N_ENT * D * 4);
  float* x2  = (float*)take((size_t)N_ENT * D * 4);
  float* r1  = (float*)take((size_t)NREL2 * D * 4);
  float* r2  = (float*)take((size_t)NREL2 * D * 4);
  float* statM = (float*)take(256 * 4);
  float* statR = (float*)take(256 * 4);
  float* st0   = (float*)take(64);
  const size_t fragD = (size_t)KT_D * NT_D * 512;     // 91 frags per DxD weight
  _Float16* p_w1in   = (_Float16*)take(fragD * 2);
  _Float16* p_w1out  = (_Float16*)take(fragD * 2);
  _Float16* p_w1loop = (_Float16*)take(fragD * 2);
  _Float16* p_w1rel  = (_Float16*)take(fragD * 2);
  _Float16* p_w2in   = (_Float16*)take(fragD * 2);
  _Float16* p_w2out  = (_Float16*)take(fragD * 2);
  _Float16* p_w2loop = (_Float16*)take(fragD * 2);
  _Float16* p_w2rel  = (_Float16*)take(fragD * 2);
  _Float16* p_fc = (_Float16*)take((size_t)KT_FC * NT_D * 512 * 2);
  _Float16* p_xT = (_Float16*)take((size_t)KT_D * NT_ENT * 512 * 2);
  float*    stack = (float*)take((size_t)BATCH * 2 * D * 4);
  _Float16* cact  = (_Float16*)take((size_t)BATCH * FLATK * 2);
  float*    hfc   = (float*)take((size_t)BATCH * D * 4);
  _Float16* hact  = (_Float16*)take((size_t)BATCH * KPAD * 2);

  // ---- pack weights (f32 -> f16 fragments) ----
  {
    int frags = KT_D * NT_D, grid = (frags + 3) / 4;
    pack_w_kernel<<<grid, 128, 0, stream>>>(w1_in,   D, D, KT_D, NT_D, p_w1in);
    pack_w_kernel<<<grid, 128, 0, stream>>>(w1_out,  D, D, KT_D, NT_D, p_w1out);
    pack_w_kernel<<<grid, 128, 0, stream>>>(w1_loop, D, D, KT_D, NT_D, p_w1loop);
    pack_w_kernel<<<grid, 128, 0, stream>>>(w1_rel,  D, D, KT_D, NT_D, p_w1rel);
    pack_w_kernel<<<grid, 128, 0, stream>>>(w2_in,   D, D, KT_D, NT_D, p_w2in);
    pack_w_kernel<<<grid, 128, 0, stream>>>(w2_out,  D, D, KT_D, NT_D, p_w2out);
    pack_w_kernel<<<grid, 128, 0, stream>>>(w2_loop, D, D, KT_D, NT_D, p_w2loop);
    pack_w_kernel<<<grid, 128, 0, stream>>>(w2_rel,  D, D, KT_D, NT_D, p_w2rel);
    int fragsFC = KT_FC * NT_D;
    pack_w_kernel<<<(fragsFC + 3) / 4, 128, 0, stream>>>(fc_w, FLATK, D, KT_FC, NT_D, p_fc);
  }

  const int gridMD = (N_ENT + MBLK - 1) / MBLK;   // 1563
  const int gridR  = (NREL2 + MBLK - 1) / MBLK;   // 8
  const int gridE  = E_TOT / MBLK;                // 6250

  // ======== CompGCN layer 1 ========
  hipMemsetAsync(agg, 0, (size_t)N_ENT * D * 4, stream);
  edge_msg_kernel<<<gridE, 256, 0, stream>>>(init_embed, init_rel, edge_src, edge_dst,
                                             edge_type, edge_norm, p_w1in, p_w1out, agg);
  gemm_stageA_kernel<<<gridMD, 256, 0, stream>>>(init_embed, N_ENT, D, w1_lrel, p_w1loop, NT_D,
                                                 agg, b1, 1.f / 3.f, agg, D, D);
  colstats_kernel<<<D, 256, 0, stream>>>(agg, N_ENT, D, statM, statR);
  bn_tanh_kernel<<<8192, 256, 0, stream>>>(agg, statM, statR, x1, (size_t)N_ENT * D, D);
  gemm_stageA_kernel<<<gridR, 256, 0, stream>>>(init_rel, NREL2, D, nullptr, p_w1rel, NT_D,
                                                nullptr, nullptr, 1.f, r1, D, D);

  // ======== CompGCN layer 2 ========
  hipMemsetAsync(agg, 0, (size_t)N_ENT * D * 4, stream);
  edge_msg_kernel<<<gridE, 256, 0, stream>>>(x1, r1, edge_src, edge_dst,
                                             edge_type, edge_norm, p_w2in, p_w2out, agg);
  gemm_stageA_kernel<<<gridMD, 256, 0, stream>>>(x1, N_ENT, D, w2_lrel, p_w2loop, NT_D,
                                                 agg, b2, 1.f / 3.f, agg, D, D);
  colstats_kernel<<<D, 256, 0, stream>>>(agg, N_ENT, D, statM, statR);
  bn_tanh_kernel<<<8192, 256, 0, stream>>>(agg, statM, statR, x2, (size_t)N_ENT * D, D);
  gemm_stageA_kernel<<<gridR, 256, 0, stream>>>(r1, NREL2, D, nullptr, p_w2rel, NT_D,
                                                nullptr, nullptr, 1.f, r2, D, D);

  // ======== ConvE ========
  gather_stack_kernel<<<(BATCH * 2 * D + 255) / 256, 256, 0, stream>>>(x2, r2, subj, rel, stack);
  bn0_stats_kernel<<<1, 1024, 0, stream>>>(stack, st0);
  conv_kernel<<<BATCH, 256, 0, stream>>>(stack, st0, conv_k, conv_b, cact);
  chanstats_kernel<<<NFILT, 256, 0, stream>>>(cact, statM, statR);
  bn1_relu_kernel<<<8192, 256, 0, stream>>>(cact, statM, statR);
  // fc: [1024 x 39200] @ [39200 x 200]
  {
    dim3 grid(BATCH / MBLK, (NT_D + 7) / 8);
    gemm_f16A_kernel<<<grid, 256, 0, stream>>>(cact, FLATK, KT_FC, p_fc, NT_D,
                                               fc_b, hfc, D, D, 0);
  }
  colstats_kernel<<<D, 256, 0, stream>>>(hfc, BATCH, D, statM, statR);
  bn2_relu_pack_kernel<<<(BATCH * KPAD + 255) / 256, 256, 0, stream>>>(hfc, statM, statR, hact);

  // ======== logits = sigmoid(h @ x2^T + ent_bias) ========
  {
    int frags = KT_D * NT_ENT;
    pack_xT_kernel<<<(frags + 3) / 4, 128, 0, stream>>>(x2, N_ENT, D, KT_D, NT_ENT, p_xT);
    dim3 grid(BATCH / MBLK, (NT_ENT + 7) / 8);
    gemm_f16A_kernel<<<grid, 256, 0, stream>>>(hact, KPAD, KT_D, p_xT, NT_ENT,
                                               ent_bias, out, N_ENT, N_ENT, 1);
  }
}
